// JointEmbeder_83717502534012
// MI455X (gfx1250) — matrix operand order repaired and checked by reading, not compiled, not run on hardware
//
#include <hip/hip_runtime.h>
#include <hip/hip_bf16.h>

// ---------------------------------------------------------------------------
// Types for CDNA5 WMMA (wave32): V_WMMA_F32_16X16X32_BF16
// ---------------------------------------------------------------------------
typedef float  v8f    __attribute__((ext_vector_type(8)));
typedef __bf16 v16bf  __attribute__((ext_vector_type(16)));
typedef __bf16 bf16x8 __attribute__((ext_vector_type(8)));

union AFrag { v16bf v; bf16x8 h[2]; };

__device__ __forceinline__ float sigmoidf_(float x) { return 1.0f / (1.0f + __expf(-x)); }

#define BSZ   256   // batch
#define EDIM  512   // emb / hidden / lstm dim (E == H == L == 512)
#define TDESC 128
#define TNAME 6
#define TTOK  50
#define FOURH 2048

// ---------------------------------------------------------------------------
// f32 -> bf16 conversion kernels
// ---------------------------------------------------------------------------
__global__ void k_cvt(const float* __restrict__ in, __bf16* __restrict__ out, int n) {
  int i = blockIdx.x * 256 + threadIdx.x;
  if (i < n) out[i] = (__bf16)in[i];
}

__global__ void k_cvt_str(const float* __restrict__ in, __bf16* __restrict__ out,
                          int rows, int cols, int ostride, int ooff) {
  int i = blockIdx.x * 256 + threadIdx.x;
  if (i < rows * cols) {
    int r = i / cols, c = i - r * cols;
    out[(size_t)r * ostride + ooff + c] = (__bf16)in[i];
  }
}

__global__ void k_fill(float* __restrict__ p, const float* __restrict__ vp, int n) {
  int i = blockIdx.x * 256 + threadIdx.x;
  if (i < n) p[i] = vp[0];
}

// ---------------------------------------------------------------------------
// Embedding gather (optionally reverse-within-len, pack_padded semantics)
// out[b,t,:] = table[ idx[b, rev ? (t<len ? len-1-t : t) : t] ]  (bf16)
// ---------------------------------------------------------------------------
__global__ void k_gather(const int* __restrict__ idx, const int* __restrict__ lens, int rev,
                         const float* __restrict__ tab, __bf16* __restrict__ out, int T) {
  int b = blockIdx.x / T, t = blockIdx.x % T;
  int st = t;
  if (rev) { int len = lens[b]; if (t < len) st = len - 1 - t; }
  int row = idx[b * T + st];
  const float* src = tab + (size_t)row * EDIM;
  __bf16* dst = out + ((size_t)b * T + t) * EDIM;
  for (int e = threadIdx.x; e < EDIM; e += 256) dst[e] = (__bf16)src[e];
}

// tok_repr[b,e] = max_t emb_tok[tokens[b,t], e]   (bf16 out, for WMMA consumption)
__global__ void k_tokpool(const int* __restrict__ toks, const float* __restrict__ tab,
                          __bf16* __restrict__ out) {
  int b = blockIdx.x;
  for (int e = threadIdx.x; e < EDIM; e += 256) {
    float m = -3.0e38f;
    for (int t = 0; t < TTOK; ++t) {
      int row = toks[b * TTOK + t];
      m = fmaxf(m, tab[(size_t)row * EDIM + e]);
    }
    out[(size_t)b * EDIM + e] = (__bf16)m;
  }
}

// ---------------------------------------------------------------------------
// Fused LSTM: each workgroup owns 16 batch rows. h (bf16) + staged x_t live in
// LDS; c stays in registers (C/D tile layout is invariant per wave per step).
// 16 wave32 * 2 h-tiles each cover the 512 hidden units; per h-tile we keep
// four gate accumulators (i,f,g,o) and issue 4 WMMAs per A-fragment.
// k-loops are kept rolled (#pragma unroll 1) to avoid VGPR spills.
// blockDim = 512 (16 wave32), gridDim = B/16 = 16
// ---------------------------------------------------------------------------
__global__ void k_lstm(const __bf16* __restrict__ X,    // [B, T, 512]
                       const __bf16* __restrict__ Wih,  // [2048, 512] row-major (PyTorch)
                       const __bf16* __restrict__ Whh,  // [2048, 512]
                       const float*  __restrict__ bias, // [2048]
                       const int*    __restrict__ lens, // null => unused
                       __bf16*       __restrict__ feat, // [B, T, 512] or null
                       float*        __restrict__ last, // [B, 512] or null
                       int T) {
  __shared__ __align__(32) __bf16 sx[16 * EDIM];
  __shared__ __align__(32) __bf16 sh[16 * EDIM];

  const int tid  = threadIdx.x;
  const int wave = tid >> 5;    // 0..15
  const int lane = tid & 31;
  const int lh   = lane >> 4;   // which K-half this lane supplies
  const int l16  = lane & 15;
  const int m0   = blockIdx.x * 16;

  for (int i = tid; i < 16 * EDIM; i += 512) sh[i] = (__bf16)0.0f;
  float creg[2][8];
#pragma unroll
  for (int a = 0; a < 2; ++a)
#pragma unroll
    for (int r = 0; r < 8; ++r) creg[a][r] = 0.0f;
  __syncthreads();

  for (int t = 0; t < T; ++t) {
    // stage x_t for our 16 batch rows: 16 rows * 512 bf16 = 1024 chunks of 16B
    for (int c = tid; c < 1024; c += 512) {
      int r = c >> 6, off = (c & 63) << 3;
      *(bf16x8*)(sx + r * EDIM + off) =
          *(const bf16x8*)(X + (((size_t)(m0 + r)) * T + t) * EDIM + off);
    }
    __syncthreads();   // x_t staged; previous step's h writes also visible

    __bf16 hout[2][8];
#pragma unroll
    for (int hl = 0; hl < 2; ++hl) {
      const int ht = wave * 2 + hl;          // h-tile 0..31
      const int n  = ht * 16 + l16;          // hidden unit column
      const float bi = bias[n], bf = bias[512 + n], bg = bias[1024 + n], bo = bias[1536 + n];
      v8f zi, zf, zg, zo;
#pragma unroll
      for (int r = 0; r < 8; ++r) { zi[r] = bi; zf[r] = bf; zg[r] = bg; zo[r] = bo; }

      const __bf16* wi0 = Wih + (size_t)(n) * EDIM;
      const __bf16* wi1 = Wih + (size_t)(512 + n) * EDIM;
      const __bf16* wi2 = Wih + (size_t)(1024 + n) * EDIM;
      const __bf16* wi3 = Wih + (size_t)(1536 + n) * EDIM;
      const __bf16* wh0 = Whh + (size_t)(n) * EDIM;
      const __bf16* wh1 = Whh + (size_t)(512 + n) * EDIM;
      const __bf16* wh2 = Whh + (size_t)(1024 + n) * EDIM;
      const __bf16* wh3 = Whh + (size_t)(1536 + n) * EDIM;

      // phase 1: x_t @ Wih^T   (rolled: 1 A-frag + 4 B-frags live per iter)
#pragma unroll 1
      for (int k0 = 0; k0 < EDIM; k0 += 32) {
        AFrag a;
        a.h[0] = *(const bf16x8*)(sx + l16 * EDIM + k0 + 16 * lh);
        a.h[1] = *(const bf16x8*)(sx + l16 * EDIM + k0 + 16 * lh + 8);
        v16bf b0 = *(const v16bf*)(wi0 + k0 + 16 * lh);
        v16bf b1 = *(const v16bf*)(wi1 + k0 + 16 * lh);
        v16bf b2 = *(const v16bf*)(wi2 + k0 + 16 * lh);
        v16bf b3 = *(const v16bf*)(wi3 + k0 + 16 * lh);
        zi = __builtin_amdgcn_wmma_f32_16x16x32_bf16(false, a.v, false, b0, (short)0, zi, false, false);
        zf = __builtin_amdgcn_wmma_f32_16x16x32_bf16(false, a.v, false, b1, (short)0, zf, false, false);
        zg = __builtin_amdgcn_wmma_f32_16x16x32_bf16(false, a.v, false, b2, (short)0, zg, false, false);
        zo = __builtin_amdgcn_wmma_f32_16x16x32_bf16(false, a.v, false, b3, (short)0, zo, false, false);
      }
      // phase 2: h_{t-1} @ Whh^T
#pragma unroll 1
      for (int k0 = 0; k0 < EDIM; k0 += 32) {
        AFrag a;
        a.h[0] = *(const bf16x8*)(sh + l16 * EDIM + k0 + 16 * lh);
        a.h[1] = *(const bf16x8*)(sh + l16 * EDIM + k0 + 16 * lh + 8);
        v16bf b0 = *(const v16bf*)(wh0 + k0 + 16 * lh);
        v16bf b1 = *(const v16bf*)(wh1 + k0 + 16 * lh);
        v16bf b2 = *(const v16bf*)(wh2 + k0 + 16 * lh);
        v16bf b3 = *(const v16bf*)(wh3 + k0 + 16 * lh);
        zi = __builtin_amdgcn_wmma_f32_16x16x32_bf16(false, a.v, false, b0, (short)0, zi, false, false);
        zf = __builtin_amdgcn_wmma_f32_16x16x32_bf16(false, a.v, false, b1, (short)0, zf, false, false);
        zg = __builtin_amdgcn_wmma_f32_16x16x32_bf16(false, a.v, false, b2, (short)0, zg, false, false);
        zo = __builtin_amdgcn_wmma_f32_16x16x32_bf16(false, a.v, false, b3, (short)0, zo, false, false);
      }

      // elementwise gate math (PyTorch order i,f,g,o); c stays in VGPRs
#pragma unroll
      for (int r = 0; r < 8; ++r) {
        float iv = sigmoidf_(zi[r]);
        float fv = sigmoidf_(zf[r]);
        float gv = tanhf(zg[r]);
        float ov = sigmoidf_(zo[r]);
        float c  = fv * creg[hl][r] + iv * gv;
        creg[hl][r] = c;
        float h = ov * tanhf(c);
        hout[hl][r] = (__bf16)h;
        int m = 8 * lh + r;
        if (feat) feat[(((size_t)(m0 + m)) * T + t) * EDIM + n] = hout[hl][r];
        if (last && lens && t == lens[m0 + m] - 1) last[(size_t)(m0 + m) * EDIM + n] = h;
      }
    }
    __syncthreads();   // all waves done reading sh for this step
#pragma unroll
    for (int hl = 0; hl < 2; ++hl) {
      const int n = (wave * 2 + hl) * 16 + l16;
#pragma unroll
      for (int r = 0; r < 8; ++r) sh[(8 * lh + r) * EDIM + n] = hout[hl][r];
    }
    // next iteration's stage+barrier orders these writes before the next reads
  }
}

// ---------------------------------------------------------------------------
// Generic WMMA GEMM: out = act( A[M,K] @ Bw[N,K]^T + bias + Cprev )
// One 16x16 output tile per wave; 8 waves per block.
// mode 0: f32 out        mode 1: tanh -> bf16 out
// mode 2: attention score fusion: score[m] += sum_n tanh(v) * svec[n]
// ---------------------------------------------------------------------------
__global__ void k_gemm(const __bf16* __restrict__ A, const __bf16* __restrict__ Bw,
                       const float* __restrict__ bias, const float* __restrict__ Cprev,
                       float* __restrict__ outF, __bf16* __restrict__ outB,
                       float* __restrict__ score, const float* __restrict__ svec,
                       int M, int N, int K, int mode) {
  const int wave = threadIdx.x >> 5;
  const int lane = threadIdx.x & 31;
  const int lh   = lane >> 4;
  const int l16  = lane & 15;

  const int tiles_n = N >> 4;
  const int total   = (M >> 4) * tiles_n;
  const int tile    = blockIdx.x * 8 + wave;
  if (tile >= total) return;                     // wave-uniform exit
  const int mt = tile / tiles_n;
  const int nt = tile - mt * tiles_n;

  const int n  = nt * 16 + l16;
  const int am = mt * 16 + l16;
  const __bf16* arow = A  + (size_t)am * K;
  const __bf16* brow = Bw + (size_t)n  * K;

  v8f acc;
  const float bv = bias ? bias[n] : 0.0f;
#pragma unroll
  for (int r = 0; r < 8; ++r) acc[r] = bv;

  for (int k0 = 0; k0 < K; k0 += 32) {
    __builtin_prefetch(brow + k0 + 256, 0, 1);   // global_prefetch_b8
    AFrag a;
    a.h[0] = *(const bf16x8*)(arow + k0 + 16 * lh);
    a.h[1] = *(const bf16x8*)(arow + k0 + 16 * lh + 8);
    v16bf b = *(const v16bf*)(brow + k0 + 16 * lh);
    acc = __builtin_amdgcn_wmma_f32_16x16x32_bf16(false, a.v, false, b, (short)0, acc, false, false);
  }

#pragma unroll
  for (int r = 0; r < 8; ++r) {
    const int m = mt * 16 + 8 * lh + r;
    float v = acc[r];
    if (Cprev) v += Cprev[(size_t)m * N + n];
    if (mode == 1) {
      outB[(size_t)m * N + n] = (__bf16)tanhf(v);
    } else if (mode == 2) {
      v = tanhf(v) * svec[n];
      v += __shfl_xor(v, 1);
      v += __shfl_xor(v, 2);
      v += __shfl_xor(v, 4);
      v += __shfl_xor(v, 8);                     // sum over 16 lanes of this half
      if (l16 == 0) atomicAdd(&score[m], v);
    } else {
      outF[(size_t)m * N + n] = v;
    }
  }
}

// ---------------------------------------------------------------------------
// Masked softmax over T=128 + weighted pooling + tanh.  One block per batch row.
// ---------------------------------------------------------------------------
__global__ void k_attnpool(const float* __restrict__ score, const int* __restrict__ lens,
                           const __bf16* __restrict__ feat, float* __restrict__ outv) {
  __shared__ float red[256];
  __shared__ float w[TDESC];
  const int b = blockIdx.x, tid = threadIdx.x;
  const int len = lens[b];

  float s = -3.0e38f;
  if (tid < TDESC && tid < len) s = score[b * TDESC + tid];
  red[tid] = s; __syncthreads();
  for (int o = 128; o > 0; o >>= 1) { if (tid < o) red[tid] = fmaxf(red[tid], red[tid + o]); __syncthreads(); }
  const float mx = red[0]; __syncthreads();

  float e = 0.0f;
  if (tid < TDESC && tid < len) e = __expf(s - mx);
  if (tid < TDESC) w[tid] = e;
  red[tid] = e; __syncthreads();
  for (int o = 128; o > 0; o >>= 1) { if (tid < o) red[tid] += red[tid + o]; __syncthreads(); }
  const float denom = red[0]; __syncthreads();
  if (tid < TDESC) w[tid] = w[tid] / denom;
  __syncthreads();

  for (int h = tid; h < EDIM; h += 256) {
    float acc = 0.0f;
    for (int t = 0; t < TDESC; ++t)
      acc += w[t] * (float)feat[(((size_t)b) * TDESC + t) * EDIM + h];
    outv[(size_t)b * EDIM + h] = tanhf(acc);
  }
}

// ---------------------------------------------------------------------------
// Cosine similarities + hinge loss mean.  One block, one batch row per thread.
// ---------------------------------------------------------------------------
__global__ void k_loss(const float* __restrict__ code, const float* __restrict__ da,
                       const float* __restrict__ dn, float* __restrict__ out) {
  __shared__ float red[256];
  const int b = threadIdx.x;
  const float eps = 1e-8f;
  float nc = 0, na = 0, nn = 0, dA = 0, dN = 0;
  for (int i = 0; i < EDIM; ++i) {
    float c = code[(size_t)b * EDIM + i];
    float a = da[(size_t)b * EDIM + i];
    float x = dn[(size_t)b * EDIM + i];
    nc += c * c; na += a * a; nn += x * x; dA += c * a; dN += c * x;
  }
  nc = sqrtf(nc); na = sqrtf(na); nn = sqrtf(nn);
  float as = dA / (fmaxf(nc, eps) * fmaxf(na, eps));
  float ns = dN / (fmaxf(nc, eps) * fmaxf(nn, eps));
  red[b] = fmaxf(0.6f - as + ns, 1e-6f);
  __syncthreads();
  for (int o = 128; o > 0; o >>= 1) { if (b < o) red[b] += red[b + o]; __syncthreads(); }
  if (b == 0) out[0] = red[0] / 256.0f;
}

// ---------------------------------------------------------------------------
// Host launch
// ---------------------------------------------------------------------------
extern "C" void kernel_launch(void* const* d_in, const int* in_sizes, int n_in,
                              void* d_out, int out_size, void* d_ws, size_t ws_size,
                              hipStream_t stream) {
  const int*   name     = (const int*)d_in[0];
  const int*   name_len = (const int*)d_in[1];
  const int*   tokens   = (const int*)d_in[2];
  const int*   desc_a   = (const int*)d_in[4];
  const int*   desc_al  = (const int*)d_in[5];
  const int*   desc_n   = (const int*)d_in[6];
  const int*   desc_nl  = (const int*)d_in[7];
  const float* emb_name = (const float*)d_in[8];
  const float* emb_tok  = (const float*)d_in[9];
  const float* emb_desc = (const float*)d_in[10];
  const float* nWih_f = (const float*)d_in[11];
  const float* nWhh_f = (const float*)d_in[12];
  const float* nb_f   = (const float*)d_in[13];
  const float* nWih_b = (const float*)d_in[14];
  const float* nWhh_b = (const float*)d_in[15];
  const float* nb_b   = (const float*)d_in[16];
  const float* dWih   = (const float*)d_in[17];
  const float* dWhh   = (const float*)d_in[18];
  const float* db     = (const float*)d_in[19];
  const float* wnameW = (const float*)d_in[20];
  const float* wnameB = (const float*)d_in[21];
  const float* wtokW  = (const float*)d_in[22];
  const float* wtokB  = (const float*)d_in[23];
  const float* fuse3W = (const float*)d_in[24];
  const float* fuse3B = (const float*)d_in[25];
  const float* attnW  = (const float*)d_in[26];
  const float* attnB  = (const float*)d_in[27];
  const float* attnS  = (const float*)d_in[28];
  const float* attnSB = (const float*)d_in[29];
  float* out = (float*)d_out;

  char* wp = (char*)d_ws;
  auto alloc = [&](size_t bytes) -> void* {
    void* p = (void*)wp;
    wp += (bytes + 255) & ~(size_t)255;
    return p;
  };

  const size_t WLSTM = (size_t)FOURH * EDIM;   // 2048*512
  __bf16* dWih_b  = (__bf16*)alloc(WLSTM * 2);
  __bf16* dWhh_b  = (__bf16*)alloc(WLSTM * 2);
  __bf16* nWihf_b = (__bf16*)alloc(WLSTM * 2);
  __bf16* nWhhf_b = (__bf16*)alloc(WLSTM * 2);
  __bf16* nWihb_b = (__bf16*)alloc(WLSTM * 2);
  __bf16* nWhhb_b = (__bf16*)alloc(WLSTM * 2);
  __bf16* attnW_b = (__bf16*)alloc((size_t)EDIM * EDIM * 2);
  __bf16* wnameW_b= (__bf16*)alloc((size_t)EDIM * 2 * EDIM * 2);
  __bf16* wtokW_b = (__bf16*)alloc((size_t)EDIM * EDIM * 2);
  __bf16* fuse3W_b= (__bf16*)alloc((size_t)EDIM * EDIM * 2);

  const size_t XD = (size_t)BSZ * TDESC * EDIM;
  __bf16* Xa    = (__bf16*)alloc(XD * 2);
  __bf16* Xn    = (__bf16*)alloc(XD * 2);
  __bf16* Xnf   = (__bf16*)alloc((size_t)BSZ * TNAME * EDIM * 2);
  __bf16* Xnb   = (__bf16*)alloc((size_t)BSZ * TNAME * EDIM * 2);
  __bf16* featA = (__bf16*)alloc(XD * 2);
  __bf16* featN = (__bf16*)alloc(XD * 2);
  float*  scoreA = (float*)alloc((size_t)BSZ * TDESC * 4);
  float*  scoreN = (float*)alloc((size_t)BSZ * TDESC * 4);
  float*  da     = (float*)alloc((size_t)BSZ * EDIM * 4);
  float*  dn     = (float*)alloc((size_t)BSZ * EDIM * 4);
  float*  lastF  = (float*)alloc((size_t)BSZ * EDIM * 4);
  float*  lastB  = (float*)alloc((size_t)BSZ * EDIM * 4);
  __bf16* nrepr  = (__bf16*)alloc((size_t)BSZ * 2 * EDIM * 2);
  __bf16* tokb   = (__bf16*)alloc((size_t)BSZ * EDIM * 2);
  float*  fpre   = (float*)alloc((size_t)BSZ * EDIM * 4);
  __bf16* fused  = (__bf16*)alloc((size_t)BSZ * EDIM * 2);
  float*  code   = (float*)alloc((size_t)BSZ * EDIM * 4);

  // 1. weight conversions to bf16
  const int nW = (int)WLSTM, nS = EDIM * EDIM, nNW = EDIM * 2 * EDIM;
  k_cvt<<<(nW + 255) / 256, 256, 0, stream>>>(dWih,   dWih_b,  nW);
  k_cvt<<<(nW + 255) / 256, 256, 0, stream>>>(dWhh,   dWhh_b,  nW);
  k_cvt<<<(nW + 255) / 256, 256, 0, stream>>>(nWih_f, nWihf_b, nW);
  k_cvt<<<(nW + 255) / 256, 256, 0, stream>>>(nWhh_f, nWhhf_b, nW);
  k_cvt<<<(nW + 255) / 256, 256, 0, stream>>>(nWih_b, nWihb_b, nW);
  k_cvt<<<(nW + 255) / 256, 256, 0, stream>>>(nWhh_b, nWhhb_b, nW);
  k_cvt<<<(nS + 255) / 256, 256, 0, stream>>>(attnW,  attnW_b, nS);
  k_cvt<<<(nNW + 255) / 256, 256, 0, stream>>>(wnameW, wnameW_b, nNW);
  k_cvt<<<(nS + 255) / 256, 256, 0, stream>>>(wtokW,  wtokW_b, nS);
  k_cvt<<<(nS + 255) / 256, 256, 0, stream>>>(fuse3W, fuse3W_b, nS);

  // 2. embedding gathers
  k_gather<<<BSZ * TDESC, 256, 0, stream>>>(desc_a, nullptr, 0, emb_desc, Xa, TDESC);
  k_gather<<<BSZ * TDESC, 256, 0, stream>>>(desc_n, nullptr, 0, emb_desc, Xn, TDESC);
  k_gather<<<BSZ * TNAME, 256, 0, stream>>>(name, nullptr, 0, emb_name, Xnf, TNAME);
  k_gather<<<BSZ * TNAME, 256, 0, stream>>>(name, name_len, 1, emb_name, Xnb, TNAME);
  k_tokpool<<<BSZ, 256, 0, stream>>>(tokens, emb_tok, tokb);

  // 3. LSTMs (16 workgroups x 16 waves: one 16-row batch tile per WG)
  k_lstm<<<BSZ / 16, 512, 0, stream>>>(Xa, dWih_b, dWhh_b, db, nullptr, featA, nullptr, TDESC);
  k_lstm<<<BSZ / 16, 512, 0, stream>>>(Xn, dWih_b, dWhh_b, db, nullptr, featN, nullptr, TDESC);
  k_lstm<<<BSZ / 16, 512, 0, stream>>>(Xnf, nWihf_b, nWhhf_b, nb_f, name_len, nullptr, lastF, TNAME);
  k_lstm<<<BSZ / 16, 512, 0, stream>>>(Xnb, nWihb_b, nWhhb_b, nb_b, name_len, nullptr, lastB, TNAME);

  // 4. attention: init score with attn_sb, fused GEMM+score, softmax+pool
  const int nSc = BSZ * TDESC;
  k_fill<<<(nSc + 255) / 256, 256, 0, stream>>>(scoreA, attnSB, nSc);
  k_fill<<<(nSc + 255) / 256, 256, 0, stream>>>(scoreN, attnSB, nSc);
  {
    const int M = BSZ * TDESC, N = EDIM, K = EDIM;
    const int blocks = ((M / 16) * (N / 16) + 7) / 8;
    k_gemm<<<blocks, 256, 0, stream>>>(featA, attnW_b, attnB, nullptr, nullptr, nullptr,
                                       scoreA, attnS, M, N, K, 2);
    k_gemm<<<blocks, 256, 0, stream>>>(featN, attnW_b, attnB, nullptr, nullptr, nullptr,
                                       scoreN, attnS, M, N, K, 2);
  }
  k_attnpool<<<BSZ, 256, 0, stream>>>(scoreA, desc_al, featA, da);
  k_attnpool<<<BSZ, 256, 0, stream>>>(scoreN, desc_nl, featN, dn);

  // 5. code encoding: name_repr = [lastF | lastB] (bf16), fuse, project
  const int nL = BSZ * EDIM;
  k_cvt_str<<<(nL + 255) / 256, 256, 0, stream>>>(lastF, nrepr, BSZ, EDIM, 2 * EDIM, 0);
  k_cvt_str<<<(nL + 255) / 256, 256, 0, stream>>>(lastB, nrepr, BSZ, EDIM, 2 * EDIM, EDIM);
  {
    const int blocks = ((BSZ / 16) * (EDIM / 16) + 7) / 8;
    // fused_pre = name_repr @ w_name_W^T + w_name_b
    k_gemm<<<blocks, 256, 0, stream>>>(nrepr, wnameW_b, wnameB, nullptr, fpre, nullptr,
                                       nullptr, nullptr, BSZ, EDIM, 2 * EDIM, 0);
    // fused = tanh(tok @ w_tok_W^T + w_tok_b + fused_pre)   -> bf16
    k_gemm<<<blocks, 256, 0, stream>>>(tokb, wtokW_b, wtokB, fpre, nullptr, fused,
                                       nullptr, nullptr, BSZ, EDIM, EDIM, 1);
    // code = fused @ fuse3_W^T + fuse3_b
    k_gemm<<<blocks, 256, 0, stream>>>(fused, fuse3W_b, fuse3B, nullptr, code, nullptr,
                                       nullptr, nullptr, BSZ, EDIM, EDIM, 0);
  }

  // 6. cosine + hinge loss
  k_loss<<<1, 256, 0, stream>>>(code, da, dn, out);

  (void)in_sizes; (void)n_in; (void)out_size; (void)ws_size;
}
